// Branch_7052336300565
// MI455X (gfx1250) — matrix operand order repaired and checked by reading, not compiled, not run on hardware
//
#include <hip/hip_runtime.h>

typedef __attribute__((ext_vector_type(16))) __bf16          v16bf;
typedef __attribute__((ext_vector_type(8)))  float           v8f;
typedef __attribute__((ext_vector_type(8)))  unsigned short  u16x8;

union ABFrag { v16bf bf; u16x8 us[2]; };

constexpr int Bn = 8192, Dn = 512, Cn = 1000, Kn = 8;
constexpr float EPSf = 1e-8f, TEMPf = 0.2f, NEGf = -10000.0f;

__device__ __forceinline__ unsigned short f2bf(float f) {
  union { float f; unsigned u; } v; v.f = f;
  unsigned u = v.u;
  u += 0x7fffu + ((u >> 16) & 1u);   // round-to-nearest-even
  return (unsigned short)(u >> 16);
}

__device__ __forceinline__ float wred_sum(float v) {
  #pragma unroll
  for (int o = 16; o; o >>= 1) v += __shfl_xor(v, o, 32);
  return v;
}
__device__ __forceinline__ float wred_max(float v) {
  #pragma unroll
  for (int o = 16; o; o >>= 1) v = fmaxf(v, __shfl_xor(v, o, 32));
  return v;
}

// ---- precision/layout prep kernels -----------------------------------------
__global__ __launch_bounds__(256) void k_cast(const float* __restrict__ in,
                                              unsigned short* __restrict__ out, int n) {
  int i = blockIdx.x * 256 + threadIdx.x;
  if (i < n) out[i] = f2bf(in[i]);
}

// W1 [K][d][e] -> W1t [K][e][d] (bf16), so GEMM B-fragments are K-contiguous
__global__ __launch_bounds__(256) void k_tw1(const float* __restrict__ in,
                                             unsigned short* __restrict__ out) {
  int i = blockIdx.x * 256 + threadIdx.x;           // K*D*D total, exact grid
  int e = i % Dn, d = (i / Dn) % Dn, k = i / (Dn * Dn);
  out[((size_t)k * Dn + e) * Dn + d] = f2bf(in[i]);
}

// W2 [K][e][c] -> W2t [K][c][e]
__global__ __launch_bounds__(256) void k_tw2(const float* __restrict__ in,
                                             unsigned short* __restrict__ out) {
  int i = blockIdx.x * 256 + threadIdx.x;           // K*D*C total, exact grid
  int c = i % Cn, e = (i / Cn) % Dn, k = i / (Dn * Cn);
  out[((size_t)k * Cn + c) * Dn + e] = f2bf(in[i]);
}

// ---- GEMM1: h[k][b][e] = relu(z @ W1[k] + b1[k]), bf16 out -----------------
// block = 8 waves; wave tile = 16(M) x 64(N); grid (B/128, D/64, K)
// double-buffered register pipeline: prefetch step k+1 while WMMA-ing step k
__global__ __launch_bounds__(256) void k_gemm1(const unsigned short* __restrict__ zb,
                                               const unsigned short* __restrict__ w1t,
                                               const float* __restrict__ b1,
                                               unsigned short* __restrict__ hb) {
  const int wave = threadIdx.x >> 5, lane = threadIdx.x & 31;
  const int hi = lane >> 4, lr = lane & 15;
  const int m0 = blockIdx.x * 128 + wave * 16;
  const int n0 = blockIdx.y * 64;
  const int k  = blockIdx.z;

  const unsigned short* Ap = zb + (size_t)(m0 + lr) * Dn + hi * 16;
  const unsigned short* Wk = w1t + (size_t)k * Dn * Dn;
  const unsigned short* Wrow[4];
  #pragma unroll
  for (int s = 0; s < 4; ++s)
    Wrow[s] = Wk + (size_t)(n0 + s * 16 + lr) * Dn + hi * 16;

  ABFrag a[2], bfr[2][4];
  a[0].us[0] = *(const u16x8*)(Ap);
  a[0].us[1] = *(const u16x8*)(Ap + 8);
  #pragma unroll
  for (int s = 0; s < 4; ++s) {
    bfr[0][s].us[0] = *(const u16x8*)(Wrow[s]);
    bfr[0][s].us[1] = *(const u16x8*)(Wrow[s] + 8);
  }

  v8f acc[4] = {};
  #pragma unroll
  for (int kk = 0; kk < Dn; kk += 32) {
    const int cur = (kk >> 5) & 1, nxt = cur ^ 1;
    if (kk + 32 < Dn) {                       // prefetch next K-step
      a[nxt].us[0] = *(const u16x8*)(Ap + kk + 32);
      a[nxt].us[1] = *(const u16x8*)(Ap + kk + 40);
      #pragma unroll
      for (int s = 0; s < 4; ++s) {
        bfr[nxt][s].us[0] = *(const u16x8*)(Wrow[s] + kk + 32);
        bfr[nxt][s].us[1] = *(const u16x8*)(Wrow[s] + kk + 40);
      }
    }
    #pragma unroll
    for (int s = 0; s < 4; ++s)
      acc[s] = __builtin_amdgcn_wmma_f32_16x16x32_bf16(
          false, a[cur].bf, false, bfr[cur][s].bf, (short)0, acc[s], false, false);
  }

  #pragma unroll
  for (int s = 0; s < 4; ++s) {
    int col = n0 + s * 16 + lr;
    float bias = b1[k * Dn + col];
    #pragma unroll
    for (int r = 0; r < 8; ++r) {
      int row = m0 + r + hi * 8;
      float v = acc[s][r] + bias;
      hb[((size_t)k * Bn + row) * Dn + col] = f2bf(v > 0.f ? v : 0.f);
    }
  }
}

// ---- GEMM2: logits_all[b][k][c] = h[k][b] @ W2[k] + b2[k], f32 out ---------
// grid (B/128, 16, K); N covers 1024, guarded to C=1000
__global__ __launch_bounds__(256) void k_gemm2(const unsigned short* __restrict__ hb,
                                               const unsigned short* __restrict__ w2t,
                                               const float* __restrict__ b2,
                                               float* __restrict__ L) {
  const int wave = threadIdx.x >> 5, lane = threadIdx.x & 31;
  const int hi = lane >> 4, lr = lane & 15;
  const int m0 = blockIdx.x * 128 + wave * 16;
  const int n0 = blockIdx.y * 64;
  const int k  = blockIdx.z;

  const unsigned short* Ap = hb + ((size_t)k * Bn + m0 + lr) * Dn + hi * 16;
  const unsigned short* Wk = w2t + (size_t)k * Cn * Dn;
  const u16x8 zv = {};

  const unsigned short* Wrow[4];
  bool valid[4];
  #pragma unroll
  for (int s = 0; s < 4; ++s) {
    int col = n0 + s * 16 + lr;
    valid[s] = (col < Cn);
    Wrow[s] = Wk + (size_t)(valid[s] ? col : 0) * Dn + hi * 16;
  }

  ABFrag a[2], bfr[2][4];
  a[0].us[0] = *(const u16x8*)(Ap);
  a[0].us[1] = *(const u16x8*)(Ap + 8);
  #pragma unroll
  for (int s = 0; s < 4; ++s) {
    if (valid[s]) {
      bfr[0][s].us[0] = *(const u16x8*)(Wrow[s]);
      bfr[0][s].us[1] = *(const u16x8*)(Wrow[s] + 8);
    } else {
      bfr[0][s].us[0] = zv; bfr[0][s].us[1] = zv;
    }
  }

  v8f acc[4] = {};
  #pragma unroll
  for (int kk = 0; kk < Dn; kk += 32) {
    const int cur = (kk >> 5) & 1, nxt = cur ^ 1;
    if (kk + 32 < Dn) {                       // prefetch next K-step
      a[nxt].us[0] = *(const u16x8*)(Ap + kk + 32);
      a[nxt].us[1] = *(const u16x8*)(Ap + kk + 40);
      #pragma unroll
      for (int s = 0; s < 4; ++s) {
        if (valid[s]) {
          bfr[nxt][s].us[0] = *(const u16x8*)(Wrow[s] + kk + 32);
          bfr[nxt][s].us[1] = *(const u16x8*)(Wrow[s] + kk + 40);
        } else {
          bfr[nxt][s].us[0] = zv; bfr[nxt][s].us[1] = zv;
        }
      }
    }
    #pragma unroll
    for (int s = 0; s < 4; ++s)
      acc[s] = __builtin_amdgcn_wmma_f32_16x16x32_bf16(
          false, a[cur].bf, false, bfr[cur][s].bf, (short)0, acc[s], false, false);
  }

  #pragma unroll
  for (int s = 0; s < 4; ++s) {
    int col = n0 + s * 16 + lr;
    if (col >= Cn) continue;
    float bias = b2[k * Cn + col];
    #pragma unroll
    for (int r = 0; r < 8; ++r) {
      int row = m0 + r + hi * 8;
      L[((size_t)row * Kn + k) * Cn + col] = acc[s][r] + bias;
    }
  }
}

// ---- gating + mix: one wave per sample -------------------------------------
__global__ __launch_bounds__(256) void k_gate(const float* __restrict__ L,
                                              const int* __restrict__ n_exp,
                                              float* __restrict__ out) {
  const int lane = threadIdx.x & 31;
  const int b = blockIdx.x * 8 + (threadIdx.x >> 5);
  const float* Lb = L + (size_t)b * Kn * Cn;

  // pass 1: per-expert max logit
  float mx[Kn];
  #pragma unroll
  for (int k = 0; k < Kn; ++k) mx[k] = -INFINITY;
  for (int c = lane; c < Cn; c += 32) {
    #pragma unroll
    for (int k = 0; k < Kn; ++k) mx[k] = fmaxf(mx[k], Lb[k * Cn + c]);
  }
  #pragma unroll
  for (int k = 0; k < Kn; ++k) mx[k] = wred_max(mx[k]);

  // pass 2: softmax denominators + all 36 pairwise exp-products
  float s1[Kn]; float pp[36];
  #pragma unroll
  for (int k = 0; k < Kn; ++k) s1[k] = 0.f;
  #pragma unroll
  for (int t = 0; t < 36; ++t) pp[t] = 0.f;
  for (int c = lane; c < Cn; c += 32) {
    float e[Kn];
    #pragma unroll
    for (int k = 0; k < Kn; ++k) {
      e[k] = __expf(Lb[k * Cn + c] - mx[k]);
      s1[k] += e[k];
    }
    int idx = 0;
    #pragma unroll
    for (int i = 0; i < Kn; ++i)
      #pragma unroll
      for (int j = i; j < Kn; ++j) pp[idx++] += e[i] * e[j];
  }
  #pragma unroll
  for (int k = 0; k < Kn; ++k) s1[k] = wred_sum(s1[k]);
  {
    int idx = 0;
    #pragma unroll
    for (int i = 0; i < Kn; ++i)
      #pragma unroll
      for (int j = i; j < Kn; ++j) { pp[idx] = wred_sum(pp[idx]); ++idx; }
  }

  // conf_k = max softmax prob = exp(0)/sum = 1/s1_k ; ||p_k|| = sqrt(Pkk)/s1_k
  float conf[Kn], nrm[Kn], S[Kn][Kn];
  #pragma unroll
  for (int k = 0; k < Kn; ++k) conf[k] = 1.f / s1[k];
  {
    int idx = 0;
    #pragma unroll
    for (int i = 0; i < Kn; ++i)
      #pragma unroll
      for (int j = i; j < Kn; ++j) { S[i][j] = pp[idx]; S[j][i] = pp[idx]; ++idx; }
  }
  #pragma unroll
  for (int k = 0; k < Kn; ++k) nrm[k] = sqrtf(S[k][k]) / s1[k];
  #pragma unroll
  for (int i = 0; i < Kn; ++i)
    #pragma unroll
    for (int j = 0; j < Kn; ++j)
      S[i][j] = (S[i][j] / (s1[i] * s1[j])) / ((nrm[i] + EPSf) * (nrm[j] + EPSf));

  // greedy diverse selection (first-max tie-breaks, matching jnp.argmax)
  const int n = n_exp[b];
  int bi = 0; float bv = conf[0];
  #pragma unroll
  for (int k = 1; k < Kn; ++k) if (conf[k] > bv) { bv = conf[k]; bi = k; }
  unsigned sel = 1u << bi;
  for (int t = 1; t < Kn; ++t) {
    float bd = -INFINITY; int bj = 0;
    #pragma unroll
    for (int i = 0; i < Kn; ++i) {
      float ms = -INFINITY;
      #pragma unroll
      for (int j = 0; j < Kn; ++j)
        if ((sel >> j) & 1) ms = fmaxf(ms, S[i][j]);
      float dist = ((sel >> i) & 1) ? -INFINITY : 1.0f - ms;
      if (dist > bd) { bd = dist; bj = i; }
    }
    if (t < n) sel |= 1u << bj;
  }

  // gate softmax
  float g[Kn]; float gm = -INFINITY;
  #pragma unroll
  for (int k = 0; k < Kn; ++k) {
    g[k] = (((sel >> k) & 1) ? conf[k] : NEGf) * (1.0f / TEMPf);
    gm = fmaxf(gm, g[k]);
  }
  float gs = 0.f;
  #pragma unroll
  for (int k = 0; k < Kn; ++k) { g[k] = __expf(g[k] - gm); gs += g[k]; }
  float ginv = 1.f / gs;
  #pragma unroll
  for (int k = 0; k < Kn; ++k) g[k] *= ginv;

  if (lane < Kn) out[(size_t)Bn * Cn + (size_t)b * Kn + lane] = g[lane];

  // pass 3: gated mix
  for (int c = lane; c < Cn; c += 32) {
    float acc = 0.f;
    #pragma unroll
    for (int k = 0; k < Kn; ++k) acc += Lb[k * Cn + c] * g[k];
    out[(size_t)b * Cn + c] = acc;
  }
}

// ---- launcher ---------------------------------------------------------------
extern "C" void kernel_launch(void* const* d_in, const int* in_sizes, int n_in,
                              void* d_out, int out_size, void* d_ws, size_t ws_size,
                              hipStream_t stream) {
  (void)in_sizes; (void)n_in; (void)out_size; (void)ws_size;
  const float* z    = (const float*)d_in[0];
  const int*   nexp = (const int*)  d_in[1];
  const float* W1   = (const float*)d_in[2];
  const float* b1   = (const float*)d_in[3];
  const float* W2   = (const float*)d_in[4];
  const float* b2   = (const float*)d_in[5];
  float* out = (float*)d_out;

  char* ws = (char*)d_ws;                       // workspace layout (bytes)
  unsigned short* zb  = (unsigned short*)(ws + 0);          //  8,388,608  z bf16
  unsigned short* w1t = (unsigned short*)(ws + 8388608);    //  4,194,304  W1^T bf16
  unsigned short* w2t = (unsigned short*)(ws + 12582912);   //  8,192,000  W2^T bf16
  unsigned short* hb  = (unsigned short*)(ws + 20774912);   // 67,108,864  h bf16 [K][B][D]
  float*          L   = (float*)         (ws + 87883776);   // 262,144,000 logits_all f32

  k_cast <<<(Bn * Dn) / 256,       256, 0, stream>>>(z, zb, Bn * Dn);
  k_tw1  <<<(Kn * Dn * Dn) / 256,  256, 0, stream>>>(W1, w1t);
  k_tw2  <<<(Kn * Dn * Cn) / 256,  256, 0, stream>>>(W2, w2t);
  k_gemm1<<<dim3(Bn / 128, Dn / 64, Kn), 256, 0, stream>>>(zb, w1t, b1, hb);
  k_gemm2<<<dim3(Bn / 128, 16,      Kn), 256, 0, stream>>>(hb, w2t, b2, L);
  k_gate <<<Bn / 8, 256, 0, stream>>>(L, nexp, out);
}
